// TGN_79611513798658
// MI455X (gfx1250) — matrix operand order repaired and checked by reading, not compiled, not run on hardware
//
#include <hip/hip_runtime.h>
#include <hip/hip_bf16.h>

#define D_   172
#define MSG_ 688
#define BQ   2000
#define KNB  20

#define MSGP 704    // 22*32, padded K for messages
#define MEMP 192    // 6*32,  padded K for memory
#define KINP 544    // 17*32, padded K for k_in

typedef __attribute__((ext_vector_type(16))) __bf16 v16bf;
typedef __attribute__((ext_vector_type(8)))  float  v8f;

union ABf16 { v16bf v; unsigned u[8]; };
union CF32  { v8f v; float f[8]; };

__device__ __forceinline__ unsigned short f2bf(float x) {
  unsigned u = __builtin_bit_cast(unsigned, x);
  u += 0x7FFFu + ((u >> 16) & 1u);          // round-to-nearest-even
  return (unsigned short)(u >> 16);
}
__device__ __forceinline__ unsigned pack2(float a, float b) {
  return (unsigned)f2bf(a) | ((unsigned)f2bf(b) << 16);
}

// A fragment (16x32 bf16) from LDS row pointer: two unconditional ds_load_b128,
// zeroed via cndmask when the row is invalid (kernel 2 second M-tile only).
__device__ __forceinline__ void load_a_lds16(const unsigned short* rowptr, int k0,
                                             int h, bool ok, unsigned* a) {
  const unsigned short* p = rowptr + k0 + (h << 3);
  uint4 lo = *(const uint4*)(p);
  uint4 hi = *(const uint4*)(p + 16);
  if (!ok) {
    lo.x = lo.y = lo.z = lo.w = 0u;
    hi.x = hi.y = hi.z = hi.w = 0u;
  }
  a[0] = lo.x; a[1] = lo.y; a[2] = lo.z; a[3] = lo.w;
  a[4] = hi.x; a[5] = hi.y; a[6] = hi.z; a[7] = hi.w;
}

// B fragment from pre-swizzled bf16 weights: two unconditional global b128 loads.
// Layout: swz[((tile*ksteps + ks)*32 + lane)*8 + v]
__device__ __forceinline__ void load_b_swz(const unsigned* __restrict__ tilestep,
                                           int lane, unsigned* b) {
  const uint4* p = (const uint4*)(tilestep + lane * 8);
  uint4 lo = p[0], hi = p[1];
  b[0] = lo.x; b[1] = lo.y; b[2] = lo.z; b[3] = lo.w;
  b[4] = hi.x; b[5] = hi.y; b[6] = hi.z; b[7] = hi.w;
}

__device__ __forceinline__ v8f wmma_bf16(const ABf16& A, const ABf16& B, v8f c) {
  return __builtin_amdgcn_wmma_f32_16x16x32_bf16(false, A.v, false, B.v, (short)0, c,
                                                 false, false);
}

__device__ __forceinline__ float sigmoidf_(float x) { return 1.f / (1.f + expf(-x)); }

// ---------------------------------------------------------------------------
// Prep: swizzle an f32 weight block (rows [row0, row0+nrows) of W, K = Kd) into
// zero-padded bf16 B-fragment order for V_WMMA_F32_16X16X32_BF16.
// out dword (tile, ks, lane, v) = pack2(W[row0+tile*16+(lane&15)][ks*32+2(v&3)+8*(lane>>4)+16*(v>>2)], +1)
// ---------------------------------------------------------------------------
__global__ __launch_bounds__(256) void tgn_swz_kernel(
    const float* __restrict__ W, int ldw, int Kd, int row0, int nrows,
    int ntiles, int ksteps, unsigned* __restrict__ out) {
  int idx = blockIdx.x * 256 + threadIdx.x;
  int total = ntiles * ksteps * 256;
  if (idx >= total) return;
  int v    = idx & 7;
  int lane = (idx >> 3) & 31;
  int ts   = idx >> 8;
  int ks   = ts % ksteps;
  int tile = ts / ksteps;
  int m = lane & 15, h = lane >> 4;
  int jr = tile * 16 + m;
  int k  = ks * 32 + ((v & 3) << 1) + (h << 3) + ((v >> 2) << 4);
  unsigned val = 0u;
  if (jr < nrows && k < Kd) {
    const float* rp = W + (size_t)(row0 + jr) * ldw + k;
    val = pack2(rp[0], rp[1]);
  }
  out[idx] = val;
}

// ---------------------------------------------------------------------------
// Kernel 1: GRU memory update.  11 waves; wave w owns gate-column tile
// [16w, 16w+16) and keeps all six gate accumulators in registers.
// ---------------------------------------------------------------------------
__global__ __launch_bounds__(352) void tgn_gru_kernel(
    const float* __restrict__ memory, const float* __restrict__ messages,
    const unsigned* __restrict__ swz_ih, const unsigned* __restrict__ swz_hh,
    const float* __restrict__ b_ih, const float* __restrict__ b_hh,
    float* __restrict__ mem_upd) {
  __shared__ unsigned short lds_msg[16 * MSGP];   // 22528 B
  __shared__ unsigned short lds_mem[16 * MEMP];   //  6144 B
  const int row0 = blockIdx.x * 16;
  const int tid  = threadIdx.x;

  // stage messages: 16 rows x 172 float4 -> bf16 pairs (uint2 LDS stores)
  for (int e = tid; e < 16 * 172; e += 352) {
    int r = e / 172, c4 = e - r * 172;
    float4 f = *(const float4*)(messages + (size_t)(row0 + r) * MSG_ + c4 * 4);
    uint2 d; d.x = pack2(f.x, f.y); d.y = pack2(f.z, f.w);
    *(uint2*)(lds_msg + r * MSGP + c4 * 4) = d;
  }
  // zero pad cols 688..703 (4 uint2 per row)
  for (int e = tid; e < 16 * 4; e += 352) {
    int r = e >> 2, q = e & 3;
    uint2 z; z.x = 0u; z.y = 0u;
    *(uint2*)(lds_msg + r * MSGP + MSG_ + q * 4) = z;
  }
  // stage memory: 16 rows x 43 float4
  for (int e = tid; e < 16 * 43; e += 352) {
    int r = e / 43, c4 = e - r * 43;
    float4 f = *(const float4*)(memory + (size_t)(row0 + r) * D_ + c4 * 4);
    uint2 d; d.x = pack2(f.x, f.y); d.y = pack2(f.z, f.w);
    *(uint2*)(lds_mem + r * MEMP + c4 * 4) = d;
  }
  // zero pad cols 172..191 (5 uint2 per row)
  for (int e = tid; e < 16 * 5; e += 352) {
    int r = e / 5, q = e - r * 5;
    uint2 z; z.x = 0u; z.y = 0u;
    *(uint2*)(lds_mem + r * MEMP + D_ + q * 4) = z;
  }
  __syncthreads();

  const int wave = tid >> 5, lane = tid & 31;
  const int m = lane & 15, h = lane >> 4;
  const int col = wave * 16 + m;                  // 0..175
  const bool jv = col < D_;
  const unsigned short* arow_msg = lds_msg + m * MSGP;
  const unsigned short* arow_mem = lds_mem + m * MEMP;

  CF32 a_ir, a_iz, a_in, a_hr, a_hz, a_hn;
  #pragma unroll
  for (int v = 0; v < 8; ++v) {
    a_ir.f[v] = 0.f; a_iz.f[v] = 0.f; a_in.f[v] = 0.f;
    a_hr.f[v] = 0.f; a_hz.f[v] = 0.f; a_hn.f[v] = 0.f;
  }

  // gi = messages @ w_ih.T   (22 K-steps)
  const unsigned* bi_r = swz_ih + ((size_t)(0 * 11 + wave) * 22) * 256;
  const unsigned* bi_z = swz_ih + ((size_t)(1 * 11 + wave) * 22) * 256;
  const unsigned* bi_n = swz_ih + ((size_t)(2 * 11 + wave) * 22) * 256;
  for (int ks = 0; ks < 22; ++ks) {
    ABf16 A, Br, Bz, Bn;
    load_a_lds16(arow_msg, ks * 32, h, true, A.u);
    load_b_swz(bi_r + ks * 256, lane, Br.u);
    load_b_swz(bi_z + ks * 256, lane, Bz.u);
    load_b_swz(bi_n + ks * 256, lane, Bn.u);
    a_ir.v = wmma_bf16(A, Br, a_ir.v);
    a_iz.v = wmma_bf16(A, Bz, a_iz.v);
    a_in.v = wmma_bf16(A, Bn, a_in.v);
  }
  // gh = memory @ w_hh.T     (6 K-steps)
  const unsigned* bh_r = swz_hh + ((size_t)(0 * 11 + wave) * 6) * 256;
  const unsigned* bh_z = swz_hh + ((size_t)(1 * 11 + wave) * 6) * 256;
  const unsigned* bh_n = swz_hh + ((size_t)(2 * 11 + wave) * 6) * 256;
  for (int ks = 0; ks < 6; ++ks) {
    ABf16 A, Br, Bz, Bn;
    load_a_lds16(arow_mem, ks * 32, h, true, A.u);
    load_b_swz(bh_r + ks * 256, lane, Br.u);
    load_b_swz(bh_z + ks * 256, lane, Bz.u);
    load_b_swz(bh_n + ks * 256, lane, Bn.u);
    a_hr.v = wmma_bf16(A, Br, a_hr.v);
    a_hz.v = wmma_bf16(A, Bz, a_hz.v);
    a_hn.v = wmma_bf16(A, Bn, a_hn.v);
  }

  if (jv) {
    float bir = b_ih[col],          bhr = b_hh[col];
    float biz = b_ih[col + D_],     bhz = b_hh[col + D_];
    float bin = b_ih[col + 2 * D_], bhn = b_hh[col + 2 * D_];
    #pragma unroll
    for (int v = 0; v < 8; ++v) {
      int row = row0 + v + 8 * h;
      float r  = sigmoidf_(a_ir.f[v] + bir + a_hr.f[v] + bhr);
      float z  = sigmoidf_(a_iz.f[v] + biz + a_hz.f[v] + bhz);
      float nn = tanhf(a_in.f[v] + bin + r * (a_hn.f[v] + bhn));
      float old = memory[(size_t)row * D_ + col];
      mem_upd[(size_t)row * D_ + col] = (1.f - z) * nn + z * old;
    }
  }
}

// ---------------------------------------------------------------------------
// Kernel 2: per attention row n: gather k_in, WMMA K/V projections, masked
// 2-head softmax attention, w_o, merge MLP -> emb[n].
// ---------------------------------------------------------------------------
__global__ __launch_bounds__(256) void tgn_attn_kernel(
    const float* __restrict__ mem_upd, const float* __restrict__ edge_features,
    const float* __restrict__ edge_times, const float* __restrict__ neighbor_times,
    const float* __restrict__ time_w, const float* __restrict__ time_b,
    const float* __restrict__ w_q, const unsigned* __restrict__ swz_k,
    const unsigned* __restrict__ swz_v, const float* __restrict__ w_o,
    const float* __restrict__ mw1, const float* __restrict__ mb1,
    const float* __restrict__ mw2, const float* __restrict__ mb2,
    const int* __restrict__ src_nodes, const int* __restrict__ dst_nodes,
    const int* __restrict__ neg_nodes, const int* __restrict__ neighbors,
    const int* __restrict__ nbr_eidx, float* __restrict__ emb_out) {
  __shared__ unsigned short kin[KNB * KINP];   // 20 x 544 bf16 = 21760 B
  __shared__ float kk_l[KNB * D_];
  __shared__ float vv_l[KNB * D_];
  __shared__ float qin[2 * D_];
  __shared__ float qv[D_];
  __shared__ float sc[2 * KNB];
  __shared__ float at[2 * KNB];
  __shared__ float ctx[D_];
  __shared__ float ov[D_];
  __shared__ float h1[D_];
  __shared__ int   s_nbr[KNB];
  __shared__ int   s_eid[KNB];
  __shared__ float s_dt[KNB];

  const int n = blockIdx.x;
  const int tid = threadIdx.x;

  int node;
  if (n < BQ)            node = src_nodes[n];
  else if (n < 2 * BQ)   node = dst_nodes[n - BQ];
  else                   node = neg_nodes[n - 2 * BQ];
  const float tn = edge_times[n % BQ];

  if (tid < KNB) {
    s_nbr[tid] = neighbors[n * KNB + tid];
    s_eid[tid] = nbr_eidx[n * KNB + tid];
    s_dt[tid]  = tn - neighbor_times[n * KNB + tid];
  }
  for (int c = tid; c < 2 * D_; c += 256)
    qin[c] = (c < D_) ? mem_upd[(size_t)node * D_ + c] : cosf(time_b[c - D_]);
  __syncthreads();

  // k_in: 20 rows x (3 segments x 43 float4) -> bf16 pairs
  for (int e = tid; e < KNB * 129; e += 256) {
    int r = e / 129, g = e - r * 129;
    int seg = g / 43, c4 = g - seg * 43;
    float4 f;
    if (seg == 0) {
      f = *(const float4*)(mem_upd + (size_t)s_nbr[r] * D_ + c4 * 4);
    } else if (seg == 1) {
      f = *(const float4*)(edge_features + (size_t)s_eid[r] * D_ + c4 * 4);
    } else {
      int t = c4 * 4;
      float dt = s_dt[r];
      f.x = cosf(dt * time_w[t + 0] + time_b[t + 0]);
      f.y = cosf(dt * time_w[t + 1] + time_b[t + 1]);
      f.z = cosf(dt * time_w[t + 2] + time_b[t + 2]);
      f.w = cosf(dt * time_w[t + 3] + time_b[t + 3]);
    }
    uint2 d; d.x = pack2(f.x, f.y); d.y = pack2(f.z, f.w);
    *(uint2*)(kin + r * KINP + seg * D_ + c4 * 4) = d;
  }
  // zero pad cols 516..543 (7 uint2 per row)
  for (int e = tid; e < KNB * 7; e += 256) {
    int r = e / 7, q = e - r * 7;
    uint2 z; z.x = 0u; z.y = 0u;
    *(uint2*)(kin + r * KINP + 3 * D_ + q * 4) = z;
  }
  __syncthreads();

  // q = q_in @ w_q.T  (172 x 344 GEMV)
  if (tid < D_) {
    float s = 0.f;
    for (int c = 0; c < 2 * D_; ++c) s += qin[c] * w_q[(size_t)tid * (2 * D_) + c];
    qv[tid] = s;
  }

  // kk / vv via WMMA: 2 M-tiles x 11 N-tiles x 2 matrices = 44 jobs
  const int wave = tid >> 5, lane = tid & 31;
  const int m = lane & 15, h = lane >> 4;
  for (int job = wave; job < 44; job += 8) {
    int ntile = job >> 2;
    int mtile = (job >> 1) & 1;
    int isv   = job & 1;
    const unsigned* Wz = (isv ? swz_v : swz_k) + ((size_t)ntile * 17) * 256;
    float* out = isv ? vv_l : kk_l;
    int n0 = ntile * 16;
    int j = n0 + m; bool jvv = j < D_;
    int arow = mtile * 16 + m;
    bool rok = arow < KNB;
    const unsigned short* ar = kin + (rok ? arow : (KNB - 1)) * KINP;
    CF32 acc;
    #pragma unroll
    for (int v = 0; v < 8; ++v) acc.f[v] = 0.f;
    for (int ks = 0; ks < 17; ++ks) {
      ABf16 A, B;
      load_a_lds16(ar, ks * 32, h, rok, A.u);
      load_b_swz(Wz + ks * 256, lane, B.u);
      acc.v = wmma_bf16(A, B, acc.v);
    }
    if (jvv) {
      #pragma unroll
      for (int v = 0; v < 8; ++v) {
        int r = mtile * 16 + v + 8 * h;
        if (r < KNB) out[r * D_ + j] = acc.f[v];
      }
    }
  }
  __syncthreads();

  // scores: 2 heads x 20 neighbors, dh = 86
  if (tid < 2 * KNB) {
    int hh = tid / KNB, kn = tid - hh * KNB;
    float s = 0.f;
    for (int dd = 0; dd < 86; ++dd) s += qv[hh * 86 + dd] * kk_l[kn * D_ + hh * 86 + dd];
    s *= (1.f / sqrtf(86.f));
    if (!(s_nbr[kn] > 0)) s = -1.0e9f;
    sc[tid] = s;
  }
  __syncthreads();
  if (tid < 2) {
    float mx = -3.4e38f;
    for (int k = 0; k < KNB; ++k) mx = fmaxf(mx, sc[tid * KNB + k]);
    float sum = 0.f;
    for (int k = 0; k < KNB; ++k) {
      float e = expf(sc[tid * KNB + k] - mx);
      at[tid * KNB + k] = e; sum += e;
    }
    float inv = 1.f / sum;
    for (int k = 0; k < KNB; ++k) at[tid * KNB + k] *= inv;
  }
  __syncthreads();
  if (tid < D_) {
    int hh = tid / 86;
    float s = 0.f;
    for (int k = 0; k < KNB; ++k) s += at[hh * KNB + k] * vv_l[k * D_ + tid];
    ctx[tid] = s;
  }
  __syncthreads();
  if (tid < D_) {
    float s = 0.f;
    for (int c = 0; c < D_; ++c) s += ctx[c] * w_o[(size_t)tid * D_ + c];
    ov[tid] = s;
  }
  __syncthreads();
  if (tid < D_) {
    float s = mb1[tid];
    const float* wr = mw1 + (size_t)tid * (2 * D_);
    for (int c = 0; c < D_; ++c) s += ov[c] * wr[c];
    for (int c = 0; c < D_; ++c) s += qin[c] * wr[D_ + c];
    h1[tid] = fmaxf(s, 0.f);
  }
  __syncthreads();
  if (tid < D_) {
    float s = mb2[tid];
    for (int c = 0; c < D_; ++c) s += h1[c] * mw2[(size_t)tid * D_ + c];
    emb_out[(size_t)n * D_ + tid] = s;
  }
}

// ---------------------------------------------------------------------------
// Kernel 3: affinity MLP -> prob[0..3999]
// ---------------------------------------------------------------------------
__global__ __launch_bounds__(256) void tgn_aff_kernel(
    const float* __restrict__ emb, const float* __restrict__ aw1,
    const float* __restrict__ ab1, const float* __restrict__ aw2,
    const float* __restrict__ ab2, float* __restrict__ out) {
  __shared__ float ah[D_];
  const int j = blockIdx.x;
  const int tid = threadIdx.x;
  const float* s  = emb + (size_t)(j % BQ) * D_;
  const float* dd = emb + (size_t)((j < BQ) ? (BQ + j) : (2 * BQ + (j - BQ))) * D_;
  if (tid < D_) {
    float acc = ab1[tid];
    const float* wr = aw1 + (size_t)tid * (2 * D_);
    for (int c = 0; c < D_; ++c) acc += s[c] * wr[c];
    for (int c = 0; c < D_; ++c) acc += dd[c] * wr[D_ + c];
    ah[tid] = fmaxf(acc, 0.f);
  }
  __syncthreads();
  if (tid == 0) {
    float p = ab2[0];
    for (int i = 0; i < D_; ++i) p += ah[i] * aw2[i];
    out[j] = sigmoidf_(p);
  }
}

extern "C" void kernel_launch(void* const* d_in, const int* in_sizes, int n_in,
                              void* d_out, int out_size, void* d_ws, size_t ws_size,
                              hipStream_t stream) {
  const float* memory         = (const float*)d_in[0];
  const float* messages       = (const float*)d_in[1];
  const float* edge_features  = (const float*)d_in[2];
  const float* edge_times     = (const float*)d_in[3];
  const float* neighbor_times = (const float*)d_in[4];
  const float* time_w         = (const float*)d_in[5];
  const float* time_b         = (const float*)d_in[6];
  const float* gw_ih          = (const float*)d_in[7];
  const float* gw_hh          = (const float*)d_in[8];
  const float* gb_ih          = (const float*)d_in[9];
  const float* gb_hh          = (const float*)d_in[10];
  const float* w_q            = (const float*)d_in[11];
  const float* w_k            = (const float*)d_in[12];
  const float* w_v            = (const float*)d_in[13];
  const float* w_o            = (const float*)d_in[14];
  const float* mw1            = (const float*)d_in[15];
  const float* mb1            = (const float*)d_in[16];
  const float* mw2            = (const float*)d_in[17];
  const float* mb2            = (const float*)d_in[18];
  const float* aw1            = (const float*)d_in[19];
  const float* ab1            = (const float*)d_in[20];
  const float* aw2            = (const float*)d_in[21];
  const float* ab2            = (const float*)d_in[22];
  const int*   srcn           = (const int*)d_in[23];
  const int*   dstn           = (const int*)d_in[24];
  const int*   negn           = (const int*)d_in[25];
  const int*   nbrs           = (const int*)d_in[26];
  const int*   nbre           = (const int*)d_in[27];

  char* ws = (char*)d_ws;
  float*    mem_upd = (float*)ws;                                 // 100000*172 f32
  float*    emb     = (float*)(ws + (size_t)100000 * D_ * 4);     // 6000*172 f32
  unsigned* swz_ih  = (unsigned*)(ws + (size_t)100000 * D_ * 4 + (size_t)6000 * D_ * 4);
  unsigned* swz_hh  = swz_ih + (size_t)3 * 11 * 22 * 256;         // 185856 dwords
  unsigned* swz_k   = swz_hh + (size_t)3 * 11 * 6 * 256;          //  50688 dwords
  unsigned* swz_v   = swz_k  + (size_t)11 * 17 * 256;             //  47872 dwords
  float*    out     = (float*)d_out;                              // 4000 f32

  // weight swizzle prep (tiny; L2-resident afterwards)
  for (int g = 0; g < 3; ++g) {
    tgn_swz_kernel<<<11 * 22, 256, 0, stream>>>(gw_ih, MSG_, MSG_, g * D_, D_, 11, 22,
                                                swz_ih + (size_t)g * 11 * 22 * 256);
    tgn_swz_kernel<<<11 * 6, 256, 0, stream>>>(gw_hh, D_, D_, g * D_, D_, 11, 6,
                                               swz_hh + (size_t)g * 11 * 6 * 256);
  }
  tgn_swz_kernel<<<11 * 17, 256, 0, stream>>>(w_k, 3 * D_, 3 * D_, 0, D_, 11, 17, swz_k);
  tgn_swz_kernel<<<11 * 17, 256, 0, stream>>>(w_v, 3 * D_, 3 * D_, 0, D_, 11, 17, swz_v);

  tgn_gru_kernel<<<100000 / 16, 352, 0, stream>>>(memory, messages, swz_ih, swz_hh,
                                                  gb_ih, gb_hh, mem_upd);
  tgn_attn_kernel<<<3 * BQ, 256, 0, stream>>>(mem_upd, edge_features, edge_times,
                                              neighbor_times, time_w, time_b,
                                              w_q, swz_k, swz_v, w_o, mw1, mb1, mw2, mb2,
                                              srcn, dstn, negn, nbrs, nbre, emb);
  tgn_aff_kernel<<<2 * BQ, 256, 0, stream>>>(emb, aw1, ab1, aw2, ab2, out);
}
